// Sent_Posit_Drmm_Modeler_8065948582244
// MI455X (gfx1250) — matrix operand order repaired and checked by reading, not compiled, not run on hardware
//
#include <hip/hip_runtime.h>
#include <hip/hip_bf16.h>
#include <math.h>

// ---------------------------------------------------------------------------
// Sent_Posit_Drmm_Modeler for MI455X (gfx1250, wave32, WMMA bf16 16x16x32,
// async global->LDS staging, b128 fragment loads, transposed C-side tiles)
// ---------------------------------------------------------------------------

typedef __attribute__((ext_vector_type(16))) __bf16 v16bf;
typedef __attribute__((ext_vector_type(8)))  __bf16 v8bf;
typedef __attribute__((ext_vector_type(8)))  float  v8f;
typedef __attribute__((ext_vector_type(4)))  float  v4f;

#define Dq   30
#define Qn   30
#define Nn   4000
#define Sn   100
#define SR   112      // S padded to 7*16 M-tiles
#define SLP  0.1f

__device__ __forceinline__ float leakyf(float x) { return x >= 0.0f ? x : SLP * x; }

// A fragment (16-bit 16x32, ISA 7.12.2): lanes 0-15 hold K = 0-7,16-23;
// lanes 16-31 hold K = 8-15,24-31. Two contiguous 8xbf16 runs -> 2x b128.
__device__ __forceinline__ v16bf load_a_frag(const __bf16* row, int half) {
    v8bf lo = *(const v8bf*)(row + 8 * half);
    v8bf hi = *(const v8bf*)(row + 16 + 8 * half);
    return __builtin_shufflevector(lo, hi, 0,1,2,3,4,5,6,7,8,9,10,11,12,13,14,15);
}
// B fragment (16-bit 32x16): lane holds n fixed, K = 8h..8h+7,16+8h..16+8h+7
// (same as A). Stored transposed in LDS ([n][K]) -> 2x b128.
__device__ __forceinline__ v16bf load_b_frag(const __bf16* rowN, int half) {
    v8bf lo = *(const v8bf*)(rowN + 8 * half);
    v8bf hi = *(const v8bf*)(rowN + 16 + 8 * half);
    return __builtin_shufflevector(lo, hi, 0,1,2,3,4,5,6,7,8,9,10,11,12,13,14,15);
}

// ------------------------- LDS layout (dynamic) ----------------------------
// xraw/yA/simA/simB are TRANSPOSED: [n(32)][m(112)] so C-fragment rows are
// contiguous (v8f = 2x b128) and pooling scans are unit-stride.
#define OFF_XRAW   0            // float [32][112]
#define OFF_YA     14336        // float [32][112]
#define OFF_SIMA   28672        // float [32][112]  (doubles as async x-stage)
#define OFF_SIMB   43008        // float [32][112]
#define OFF_ABF    57344        // bf16  [114][32]
#define OFF_BBF    64640        // bf16  [114][32]
#define OFF_CBF    71936        // bf16  [114][32]
#define OFF_W1B    79232        // bf16  [3][32][32]   [tap][dout][din]
#define OFF_W2B    85376        // bf16  [3][32][32]
#define OFF_QEB    91520        // bf16  [32][32]      [q][d]
#define OFF_QCB    93568        // bf16  [32][32]      [q][d]
#define OFF_B1     95616        // float [32]
#define OFF_B2     95744        // float [32]
#define OFF_RIX    95872        // float [112]
#define OFF_CN2    96320        // float [112]
#define OFF_RCX    96768        // float [112]
#define OFF_RQI    97216        // float [32]
#define OFF_RQS    97344        // float [32]
#define OFF_QWT    97472        // float [32]
#define OFF_FEAT   97600        // float [32][6]
#define OFF_LOSUM  98368        // float
#define SMEM_BYTES 98432

// ===========================================================================
// Kernel A: question-side convs, norms, softmax term weights (1 block)
// ws layout: [0:900) qconv  [900:930) q_weights  [930:960) ||qe||  [960:990) ||qconv||
// ===========================================================================
__global__ __launch_bounds__(128) void question_kernel(
    const float* __restrict__ qe, const float* __restrict__ qidf,
    const float* __restrict__ w1, const float* __restrict__ b1,
    const float* __restrict__ w2, const float* __restrict__ b2,
    const float* __restrict__ qwW, const float* __restrict__ qwb,
    float* __restrict__ ws)
{
    __shared__ float xs[Qn][Dq];
    __shared__ float y1[Qn][Dq];
    __shared__ float y2[Qn][Dq];
    __shared__ float z[Qn];
    const int tid = threadIdx.x;

    for (int i = tid; i < Qn * Dq; i += 128) ((float*)xs)[i] = qe[i];
    __syncthreads();

    for (int i = tid; i < Qn * Dq; i += 128) {
        int q = i / Dq, o = i % Dq;
        float acc = b1[o];
        #pragma unroll
        for (int k = 0; k < 3; ++k) {
            int s = q + k - 1;
            if (s >= 0 && s < Qn)
                for (int d = 0; d < Dq; ++d) acc += xs[s][d] * w1[(o * Dq + d) * 3 + k];
        }
        y1[q][o] = leakyf(acc) + xs[q][o];
    }
    __syncthreads();
    for (int i = tid; i < Qn * Dq; i += 128) {
        int q = i / Dq, o = i % Dq;
        float acc = b2[o];
        #pragma unroll
        for (int k = 0; k < 3; ++k) {
            int s = q + k - 1;
            if (s >= 0 && s < Qn)
                for (int d = 0; d < Dq; ++d) acc += y1[s][d] * w2[(o * Dq + d) * 3 + k];
        }
        y2[q][o] = leakyf(acc) + y1[q][o];
    }
    __syncthreads();

    for (int i = tid; i < Qn * Dq; i += 128) ws[i] = ((float*)y2)[i];

    if (tid < Qn) {
        float si = 0.0f, ss = 0.0f, zz = qwb[0] + qidf[tid] * qwW[Dq];
        for (int d = 0; d < Dq; ++d) {
            si += xs[tid][d] * xs[tid][d];
            ss += y2[tid][d] * y2[tid][d];
            zz += y2[tid][d] * qwW[d];
        }
        ws[930 + tid] = sqrtf(si);
        ws[960 + tid] = sqrtf(ss);
        z[tid] = zz;
    }
    __syncthreads();
    if (tid == 0) {
        float m = z[0];
        for (int q = 1; q < Qn; ++q) m = fmaxf(m, z[q]);
        float sum = 0.0f;
        for (int q = 0; q < Qn; ++q) sum += expf(z[q] - m);
        for (int q = 0; q < Qn; ++q) ws[900 + q] = expf(z[q] - m) / sum;
    }
}

// ===========================================================================
// Kernel B: one workgroup per sentence; async x-stage, WMMA convs + cos-sims
// ===========================================================================
__global__ __launch_bounds__(256) void sent_kernel(
    const float* __restrict__ doc, const float* __restrict__ sgaf,
    const float* __restrict__ qe, const float* __restrict__ conv1w,
    const float* __restrict__ conv1b, const float* __restrict__ conv2w,
    const float* __restrict__ conv2b, const float* __restrict__ ws,
    const float* __restrict__ lq1W, const float* __restrict__ lq1b,
    const float* __restrict__ lq2W, const float* __restrict__ lq2b,
    const float* __restrict__ soW, const float* __restrict__ sob,
    float* __restrict__ out_scores)
{
    extern __shared__ __align__(32) char smem[];
    float (*xrawT)[SR]  = (float (*)[SR])(smem + OFF_XRAW);   // [n][m]
    float (*yAT)[SR]    = (float (*)[SR])(smem + OFF_YA);     // [n][m]
    float (*simAT)[SR]  = (float (*)[SR])(smem + OFF_SIMA);   // [q][s]
    float (*simBT)[SR]  = (float (*)[SR])(smem + OFF_SIMB);   // [q][s]
    __bf16 (*abf)[32]   = (__bf16 (*)[32])(smem + OFF_ABF);
    __bf16 (*bbf)[32]   = (__bf16 (*)[32])(smem + OFF_BBF);
    __bf16 (*cbf)[32]   = (__bf16 (*)[32])(smem + OFF_CBF);
    __bf16 (*w1b)[32][32] = (__bf16 (*)[32][32])(smem + OFF_W1B);  // [tap][o][d]
    __bf16 (*w2b)[32][32] = (__bf16 (*)[32][32])(smem + OFF_W2B);
    __bf16 (*qeb)[32]   = (__bf16 (*)[32])(smem + OFF_QEB);        // [q][d]
    __bf16 (*qcb)[32]   = (__bf16 (*)[32])(smem + OFF_QCB);        // [q][d]
    float* bias1 = (float*)(smem + OFF_B1);
    float* bias2 = (float*)(smem + OFF_B2);
    float* rix   = (float*)(smem + OFF_RIX);
    float* cn2   = (float*)(smem + OFF_CN2);
    float* rcx   = (float*)(smem + OFF_RCX);
    float* rqi   = (float*)(smem + OFF_RQI);
    float* rqs   = (float*)(smem + OFF_RQS);
    float* qwt   = (float*)(smem + OFF_QWT);
    float (*feats)[6] = (float (*)[6])(smem + OFF_FEAT);
    float* losum = (float*)(smem + OFF_LOSUM);
    float* xstage = (float*)(smem + OFF_SIMA);   // async staging, dead until ph.1

    const int tid  = threadIdx.x;
    const int lane = tid & 31, wave = tid >> 5;
    const int half = lane >> 4, lr = lane & 15;
    const int nblk = blockIdx.x;

    // Per-wave static tiling: 7 M-tiles x 2 N-tiles; a wave owns one N-tile
    // parity so all B fragments are loop-invariant.
    const int nt = wave & 1;
    const int nq = nt * 16 + lr;
    const int mt0 = wave >> 1;          // mt0, mt0+4  (mt0==3 -> single tile)

    // ---------------- phase 0a: async x-stage + zero fills ----------------
    // Stage 100*30 f32 (12000 B = 750 x 16B) global -> LDS via the CDNA5
    // async path (GVS addressing, per-lane 16B copies, tracked by ASYNCcnt).
    {
        const float* xg = doc + (size_t)nblk * Sn * Dq;
        #pragma unroll
        for (int pass = 0; pass < 3; ++pass) {
            int i = tid + pass * 256;
            if (i < 750) {
                unsigned ldsa = (unsigned)OFF_SIMA + (unsigned)i * 16u;
                unsigned goff = (unsigned)i * 16u;
                asm volatile("global_load_async_to_lds_b128 %0, %1, %2"
                             :: "v"(ldsa), "v"(goff), "s"(xg) : "memory");
            }
        }
    }
    { float* xz = (float*)xrawT;
      for (int i = tid; i < 32 * SR; i += 256) xz[i] = 0.0f; }
    { __bf16* bz = (__bf16*)abf;                  // abf,bbf,cbf contiguous
      for (int i = tid; i < 114 * 32 * 3; i += 256) bz[i] = (__bf16)0.0f; }
    { __bf16* wz = (__bf16*)w1b;                  // w1b,w2b,qeb,qcb contiguous
      for (int i = tid; i < (3*32*32)*2 + 32*32*2; i += 256) wz[i] = (__bf16)0.0f; }
    for (int i = tid; i < 64; i += 256) bias1[i] = 0.0f;   // bias1+bias2
    for (int i = tid; i < SR; i += 256) cn2[i] = 0.0f;
    if (tid == 0) *losum = 0.0f;
    asm volatile("s_wait_asynccnt 0x0" ::: "memory");
    __syncthreads();

    // ---------------- phase 0b: repack + small loads ----------------
    for (int i = tid; i < Sn * Dq; i += 256) {
        int s = i / Dq, d = i % Dq;
        float v = xstage[i];
        xrawT[d][s]   = v;
        abf[s + 1][d] = (__bf16)v;
    }
    for (int i = tid; i < Dq * Dq * 3; i += 256) {   // w[(o*30+d)*3+k] -> [k][o][d]
        int o = i / 90, r = i % 90, d = r / 3, k = r % 3;
        w1b[k][o][d] = (__bf16)conv1w[i];
        w2b[k][o][d] = (__bf16)conv2w[i];
    }
    if (tid < Dq) { bias1[tid] = conv1b[tid]; bias2[tid] = conv2b[tid]; }
    for (int i = tid; i < Qn * Dq; i += 256) {       // B tiles, row-major [q][d]
        int q = i / Dq, d = i % Dq;
        qeb[q][d] = (__bf16)qe[i];
        qcb[q][d] = (__bf16)ws[i];
    }
    if (tid < Qn) {
        rqi[tid] = 1.0f / ws[930 + tid];
        rqs[tid] = 1.0f / ws[960 + tid];
        qwt[tid] = ws[900 + tid];
    }
    __syncthreads();

    // ---------------- phase 1: sim_insens + conv1 (WMMA) ----------------
    {
        const v16bf bqe = load_b_frag(&qeb[nq][0], half);
        const v16bf bw0 = load_b_frag(&w1b[0][nq][0], half);
        const v16bf bw1 = load_b_frag(&w1b[1][nq][0], half);
        const v16bf bw2 = load_b_frag(&w1b[2][nq][0], half);
        const float bia = bias1[nq];
        for (int mt = mt0; mt < 7; mt += 4) {
            const int m0 = mt * 16 + lr;
            const int mb = mt * 16 + 8 * half;      // first row of this lane's Cfrag
            v16bf a1 = load_a_frag(&abf[m0 + 1][0], half);  // shared: sim A, conv kb=1

            v8f c = {0.f,0.f,0.f,0.f,0.f,0.f,0.f,0.f};
            c = __builtin_amdgcn_wmma_f32_16x16x32_bf16(false, a1, false, bqe,
                                                        (short)0, c, false, false);
            *(v8f*)&simAT[nq][mb] = c;              // contiguous: 2x ds_store_b128

            v16bf a0 = load_a_frag(&abf[m0][0], half);
            v16bf a2 = load_a_frag(&abf[m0 + 2][0], half);
            v8f cc = {0.f,0.f,0.f,0.f,0.f,0.f,0.f,0.f};
            cc = __builtin_amdgcn_wmma_f32_16x16x32_bf16(false, a0, false, bw0,
                                                         (short)0, cc, false, false);
            cc = __builtin_amdgcn_wmma_f32_16x16x32_bf16(false, a1, false, bw1,
                                                         (short)0, cc, false, false);
            cc = __builtin_amdgcn_wmma_f32_16x16x32_bf16(false, a2, false, bw2,
                                                         (short)0, cc, false, false);

            v8f res = *(const v8f*)&xrawT[nq][mb];  // unconditional, pad rows are 0
            v8f yv;
            #pragma unroll
            for (int r = 0; r < 8; ++r) yv[r] = leakyf(cc[r] + bia) + res[r];
            if (mt == 6) {                          // only tile with rows >= 100
                #pragma unroll
                for (int r = 0; r < 8; ++r) yv[r] = (mb + r < Sn) ? yv[r] : 0.0f;
            }
            *(v8f*)&yAT[nq][mb] = yv;
            #pragma unroll
            for (int r = 0; r < 8; ++r) bbf[mb + r + 1][nq] = (__bf16)yv[r];
        }
    }
    if (tid < Sn) {
        float s = 0.0f;
        for (int d = 0; d < Dq; ++d) s += xrawT[d][tid] * xrawT[d][tid];
        rix[tid] = 1.0f / sqrtf(s);
    }
    __syncthreads();

    // ---------------- phase 2: conv2 (WMMA) + row-norm accumulate ----------
    {
        const v16bf bw0 = load_b_frag(&w2b[0][nq][0], half);
        const v16bf bw1 = load_b_frag(&w2b[1][nq][0], half);
        const v16bf bw2 = load_b_frag(&w2b[2][nq][0], half);
        const float bia = bias2[nq];
        for (int mt = mt0; mt < 7; mt += 4) {
            const int m0 = mt * 16 + lr;
            const int mb = mt * 16 + 8 * half;
            v16bf a0 = load_a_frag(&bbf[m0][0], half);
            v16bf a1 = load_a_frag(&bbf[m0 + 1][0], half);
            v16bf a2 = load_a_frag(&bbf[m0 + 2][0], half);
            v8f cc = {0.f,0.f,0.f,0.f,0.f,0.f,0.f,0.f};
            cc = __builtin_amdgcn_wmma_f32_16x16x32_bf16(false, a0, false, bw0,
                                                         (short)0, cc, false, false);
            cc = __builtin_amdgcn_wmma_f32_16x16x32_bf16(false, a1, false, bw1,
                                                         (short)0, cc, false, false);
            cc = __builtin_amdgcn_wmma_f32_16x16x32_bf16(false, a2, false, bw2,
                                                         (short)0, cc, false, false);

            v8f res = *(const v8f*)&yAT[nq][mb];
            v8f yv;
            #pragma unroll
            for (int r = 0; r < 8; ++r) yv[r] = leakyf(cc[r] + bia) + res[r];
            if (mt == 6) {
                #pragma unroll
                for (int r = 0; r < 8; ++r) yv[r] = (mb + r < Sn) ? yv[r] : 0.0f;
            }
            #pragma unroll
            for (int r = 0; r < 8; ++r) {
                cbf[mb + r][nq] = (__bf16)yv[r];
                atomicAdd(&cn2[mb + r], yv[r] * yv[r]);   // ds_add_f32
            }
        }
    }
    __syncthreads();

    // ---------------- phase 3: sim_sens (WMMA) + conv norms ----------------
    {
        const v16bf bqc = load_b_frag(&qcb[nq][0], half);
        for (int mt = mt0; mt < 7; mt += 4) {
            const int m0 = mt * 16 + lr;
            const int mb = mt * 16 + 8 * half;
            v16bf a = load_a_frag(&cbf[m0][0], half);
            v8f c = {0.f,0.f,0.f,0.f,0.f,0.f,0.f,0.f};
            c = __builtin_amdgcn_wmma_f32_16x16x32_bf16(false, a, false, bqc,
                                                        (short)0, c, false, false);
            *(v8f*)&simBT[nq][mb] = c;
        }
    }
    if (tid < Sn) rcx[tid] = 1.0f / sqrtf(cn2[tid]);
    __syncthreads();

    // ---------------- phase 4: top-5 pooling (vectorized, unit-stride) -----
    if (tid < 2 * Qn) {
        const int q = tid % Qn, mat = tid / Qn;
        const float rq = (mat == 0) ? rqi[q] : rqs[q];
        const float* row = (mat == 0) ? &simAT[q][0] : &simBT[q][0];
        const float* rn  = (mat == 0) ? rix : rcx;
        float t0 = -1e30f, t1 = -1e30f, t2 = -1e30f, t3 = -1e30f, t4 = -1e30f;
        int cnt = 0;
        for (int s4 = 0; s4 < Sn; s4 += 4) {
            v4f v4 = *(const v4f*)(row + s4);
            v4f n4 = *(const v4f*)(rn + s4);
            #pragma unroll
            for (int j = 0; j < 4; ++j) {
                float v = v4[j] * n4[j] * rq;
                if (mat == 0 && v > 1.0f - 1e-3f) cnt++;
                if      (v > t0) { t4 = t3; t3 = t2; t2 = t1; t1 = t0; t0 = v; }
                else if (v > t1) { t4 = t3; t3 = t2; t2 = t1; t1 = v; }
                else if (v > t2) { t4 = t3; t3 = t2; t2 = v; }
                else if (v > t3) { t4 = t3; t3 = v; }
                else if (v > t4) { t4 = v; }
            }
        }
        const float mx = t0, av = (t0 + t1 + t2 + t3 + t4) * 0.2f;
        if (mat == 0) {
            feats[q][0] = (cnt > 0) ? 1.0f : 0.0f;
            feats[q][1] = (float)(cnt > 5 ? 5 : cnt) * 0.2f;
            feats[q][2] = mx;
            feats[q][3] = av;
        } else {
            feats[q][4] = mx;
            feats[q][5] = av;
        }
    }
    __syncthreads();

    // ---------------- phase 5: tiny MLP + sentence score -------------------
    if (tid < Qn) {
        float f[6];
        #pragma unroll
        for (int i = 0; i < 6; ++i) f[i] = feats[tid][i];
        float lo = lq2b[0];
        #pragma unroll
        for (int j = 0; j < 8; ++j) {
            float h = lq1b[j];
            #pragma unroll
            for (int i = 0; i < 6; ++i) h += f[i] * lq1W[j * 6 + i];
            lo += leakyf(h) * lq2W[j];
        }
        atomicAdd(losum, lo * qwt[tid]);
    }
    __syncthreads();
    if (tid == 0) {
        const float emit = *losum / (float)Qn;
        const float* g = sgaf + (size_t)nblk * 3;
        const float z = soW[0] * g[0] + soW[1] * g[1] + soW[2] * g[2]
                      + soW[3] * emit + sob[0];
        out_scores[nblk] = 1.0f / (1.0f + expf(-z));
    }
}

// ===========================================================================
// Kernel C: max over sentence scores + doc-level linear layer (1 block)
// ===========================================================================
__global__ __launch_bounds__(256) void final_kernel(
    const float* __restrict__ scores, const float* __restrict__ dgaf,
    const float* __restrict__ flW, const float* __restrict__ flb,
    float* __restrict__ out0)
{
    __shared__ float red[256];
    const int tid = threadIdx.x;
    float m = -1e30f;
    for (int i = tid; i < Nn; i += 256) m = fmaxf(m, scores[i]);
    red[tid] = m;
    __syncthreads();
    for (int s = 128; s > 0; s >>= 1) {
        if (tid < s) red[tid] = fmaxf(red[tid], red[tid + s]);
        __syncthreads();
    }
    if (tid == 0) {
        float f = flb[0] + red[0] * flW[0];
        #pragma unroll
        for (int i = 0; i < 4; ++i) f += dgaf[i] * flW[1 + i];
        out0[0] = f;
    }
}

// ===========================================================================
extern "C" void kernel_launch(void* const* d_in, const int* in_sizes, int n_in,
                              void* d_out, int out_size, void* d_ws, size_t ws_size,
                              hipStream_t stream)
{
    const float* doc   = (const float*)d_in[0];
    const float* sgaf  = (const float*)d_in[1];
    const float* qe    = (const float*)d_in[2];
    const float* qidf  = (const float*)d_in[3];
    const float* dgaf  = (const float*)d_in[4];
    const float* c1w   = (const float*)d_in[5];
    const float* c1b   = (const float*)d_in[6];
    const float* c2w   = (const float*)d_in[7];
    const float* c2b   = (const float*)d_in[8];
    const float* qwW   = (const float*)d_in[9];
    const float* qwb   = (const float*)d_in[10];
    const float* lq1W  = (const float*)d_in[11];
    const float* lq1b  = (const float*)d_in[12];
    const float* lq2W  = (const float*)d_in[13];
    const float* lq2b  = (const float*)d_in[14];
    const float* soW   = (const float*)d_in[15];
    const float* sob   = (const float*)d_in[16];
    const float* flW   = (const float*)d_in[17];
    const float* flb   = (const float*)d_in[18];

    float* out = (float*)d_out;     // [0]=final, [1..4000]=sent_scores
    float* ws  = (float*)d_ws;      // 990 floats used

    question_kernel<<<1, 128, 0, stream>>>(qe, qidf, c1w, c1b, c2w, c2b, qwW, qwb, ws);
    sent_kernel<<<Nn, 256, SMEM_BYTES, stream>>>(doc, sgaf, qe, c1w, c1b, c2w, c2b,
                                                 ws, lq1W, lq1b, lq2W, lq2b,
                                                 soW, sob, out + 1);
    final_kernel<<<1, 256, 0, stream>>>(out + 1, dgaf, flW, flb, out);
}
